// Similarity_44169443672264
// MI455X (gfx1250) — matrix operand order repaired and checked by reading, not compiled, not run on hardware
//
#include <hip/hip_runtime.h>

typedef __attribute__((ext_vector_type(16))) __bf16 v16bf;
typedef __attribute__((ext_vector_type(8)))  float  v8f;

#define DIMK   1024
#define NROWS  4096
#define LDS_STRIDE 40   // 32 + 8 pad (bf16 elements) -> conflict-free ds_load_b128
#define KSTEPS (DIMK / 32)

__device__ __forceinline__ unsigned short f2bf(float f) {
    unsigned int u = __float_as_uint(f);
    unsigned int r = (u + 0x7FFFu + ((u >> 16) & 1u)) >> 16;  // round-to-nearest-even
    return (unsigned short)r;
}

// Async DMA: global -> LDS, 16B per lane, tracked by ASYNCcnt (no VGPR landing).
__device__ __forceinline__ void async_ld16(unsigned lds_off, const void* gaddr) {
    asm volatile("global_load_async_to_lds_b128 %0, %1, off"
                 :: "v"(lds_off), "v"(gaddr)
                 : "memory");
}

#if __has_builtin(__builtin_amdgcn_s_wait_asynccnt)
#define WAIT_ASYNC(N) __builtin_amdgcn_s_wait_asynccnt(N)
#else
#define WAIT_ASYNC(N) asm volatile("s_wait_asynccnt %0" :: "i"(N) : "memory")
#endif

// lds byte offset of a generic pointer to __shared__ (aperture puts the LDS
// offset in addr[31:0] per the ISA flat-address rules)
__device__ __forceinline__ unsigned lds_off(const void* p) {
    return (unsigned)(unsigned long long)p;
}

// One block per row (8192 blocks): convert f32 -> bf16 into workspace and
// compute exact f32 row L2 norms.
__global__ void __launch_bounds__(256)
convert_and_norm(const float* __restrict__ x, const float* __restrict__ y,
                 unsigned short* __restrict__ xb, unsigned short* __restrict__ yb,
                 float* __restrict__ nx, float* __restrict__ ny) {
    int row = blockIdx.x;
    const float* src; unsigned short* dstb; float* dstn; int r;
    if (row < NROWS) { src = x; dstb = xb; dstn = nx; r = row; }
    else             { src = y; dstb = yb; dstn = ny; r = row - NROWS; }

    const float4* p = (const float4*)(src + (size_t)r * DIMK);
    float4 v = p[threadIdx.x];                       // 256 threads * 4 = 1024
    ushort4 b;
    b.x = f2bf(v.x); b.y = f2bf(v.y); b.z = f2bf(v.z); b.w = f2bf(v.w);
    ((ushort4*)(dstb + (size_t)r * DIMK))[threadIdx.x] = b;

    float s = v.x*v.x + v.y*v.y + v.z*v.z + v.w*v.w;
    for (int off = 16; off > 0; off >>= 1) s += __shfl_down(s, off, 32);

    __shared__ float red[8];
    int lane = threadIdx.x & 31, w = threadIdx.x >> 5;
    if (lane == 0) red[w] = s;
    __syncthreads();
    if (threadIdx.x == 0) {
        float t = 0.f;
        for (int i = 0; i < 8; ++i) t += red[i];
        dstn[r] = sqrtf(t);
    }
}

// 128x128 tile per 256-thread block; 8 waves, each wave: 64x32 patch = 4x2
// WMMA 16x16 tiles; K-step 32 via v_wmma_f32_16x16x32_bf16.
// Double-buffered LDS filled by global_load_async_to_lds_b128 (ASYNCcnt).
__global__ void __launch_bounds__(256)
cosine_gemm(const unsigned short* __restrict__ xb, const unsigned short* __restrict__ yb,
            const float* __restrict__ nx, const float* __restrict__ ny,
            float* __restrict__ out) {
    __shared__ __align__(16) unsigned short lsA[2][128 * LDS_STRIDE];
    __shared__ __align__(16) unsigned short lsB[2][128 * LDS_STRIDE];

    const int tid    = threadIdx.x;
    const int lane   = tid & 31;
    const int wv     = tid >> 5;
    const int wave_m = wv & 1;     // 0..1 -> 64-row half
    const int wave_n = wv >> 1;    // 0..3 -> 32-col strip
    const int row0   = blockIdx.y * 128;
    const int col0   = blockIdx.x * 128;

    v8f zero = {};
    v8f acc[4][2];
#pragma unroll
    for (int m = 0; m < 4; ++m)
#pragma unroll
        for (int n = 0; n < 2; ++n) acc[m][n] = zero;

    const int hi   = lane >> 4;          // 0 | 1 (K-half select per ISA layout)
    const int l15  = lane & 15;          // M for A-frag, N for B-frag
    const int a_k0 = hi ? 8  : 0;        // A chunks at a_k0 and a_k0+16
    const int b_k0 = hi ? 8  : 0;        // B chunks at b_k0 and b_k0+16

    // staging: each thread DMAs 4 chunks (8 bf16 = 16B each): rows crow and
    // crow+64 of both the A and B panels.
    const int crow = tid >> 2;           // 0..63
    const int cseg = tid & 3;            // 0..3 (16B segment within 64B row)
    const int lelem = crow * LDS_STRIDE + cseg * 8;

    const unsigned la0[2] = { lds_off(&lsA[0][lelem]), lds_off(&lsA[1][lelem]) };
    const unsigned la1[2] = { lds_off(&lsA[0][lelem + 64 * LDS_STRIDE]),
                              lds_off(&lsA[1][lelem + 64 * LDS_STRIDE]) };
    const unsigned lb0[2] = { lds_off(&lsB[0][lelem]), lds_off(&lsB[1][lelem]) };
    const unsigned lb1[2] = { lds_off(&lsB[0][lelem + 64 * LDS_STRIDE]),
                              lds_off(&lsB[1][lelem + 64 * LDS_STRIDE]) };

    const unsigned short* gA0 = xb + (size_t)(row0 + crow)      * DIMK + cseg * 8;
    const unsigned short* gA1 = xb + (size_t)(row0 + crow + 64) * DIMK + cseg * 8;
    const unsigned short* gB0 = yb + (size_t)(col0 + crow)      * DIMK + cseg * 8;
    const unsigned short* gB1 = yb + (size_t)(col0 + crow + 64) * DIMK + cseg * 8;

#define ISSUE_BATCH(buf, kk)                      \
    do {                                          \
        async_ld16(la0[buf], gA0 + (kk));         \
        async_ld16(la1[buf], gA1 + (kk));         \
        async_ld16(lb0[buf], gB0 + (kk));         \
        async_ld16(lb1[buf], gB1 + (kk));         \
    } while (0)

    ISSUE_BATCH(0, 0);                            // prefetch tile 0

    for (int it = 0; it < KSTEPS; ++it) {
        const int cur = it & 1;
        if (it + 1 < KSTEPS) {
            ISSUE_BATCH(cur ^ 1, (it + 1) * 32);  // prefetch next tile
            WAIT_ASYNC(4);                        // in-order: tile `cur` landed
        } else {
            WAIT_ASYNC(0);
        }
        __syncthreads();                          // all waves' data visible

        union { v16bf v; uint4 q[2]; } fa[4], fb[2];
#pragma unroll
        for (int m = 0; m < 4; ++m) {
            const unsigned short* base =
                &lsA[cur][(wave_m * 64 + m * 16 + l15) * LDS_STRIDE + a_k0];
            fa[m].q[0] = *(const uint4*)(base);
            fa[m].q[1] = *(const uint4*)(base + 16);
        }
#pragma unroll
        for (int n = 0; n < 2; ++n) {
            const unsigned short* base =
                &lsB[cur][(wave_n * 32 + n * 16 + l15) * LDS_STRIDE + b_k0];
            fb[n].q[0] = *(const uint4*)(base);
            fb[n].q[1] = *(const uint4*)(base + 16);
        }
#pragma unroll
        for (int m = 0; m < 4; ++m)
#pragma unroll
            for (int n = 0; n < 2; ++n)
                acc[m][n] = __builtin_amdgcn_wmma_f32_16x16x32_bf16(
                    false, fa[m].v, false, fb[n].v, (short)0, acc[m][n],
                    false, false);

        __syncthreads();                          // buffer `cur` free for reuse
    }

    // fused epilogue: out = dot * (1/TEMP) / max(nx*ny, EPS)
#pragma unroll
    for (int n = 0; n < 2; ++n) {
        const int col = col0 + wave_n * 32 + n * 16 + l15;
        const float nyv = ny[col];
#pragma unroll
        for (int m = 0; m < 4; ++m) {
#pragma unroll
            for (int r = 0; r < 8; ++r) {
                const int row = row0 + wave_m * 64 + m * 16 + hi * 8 + r;
                const float denom = fmaxf(nx[row] * nyv, 1e-8f);
                out[(size_t)row * NROWS + col] = acc[m][n][r] * (20.0f / denom);
            }
        }
    }
}

extern "C" void kernel_launch(void* const* d_in, const int* in_sizes, int n_in,
                              void* d_out, int out_size, void* d_ws, size_t ws_size,
                              hipStream_t stream) {
    const float* x = (const float*)d_in[0];
    const float* y = (const float*)d_in[1];
    float* out = (float*)d_out;

    // workspace: [Xbf16 8MB][Ybf16 8MB][nx 16KB][ny 16KB]
    unsigned short* xb = (unsigned short*)d_ws;
    unsigned short* yb = xb + (size_t)NROWS * DIMK;
    float* nx = (float*)(yb + (size_t)NROWS * DIMK);
    float* ny = nx + NROWS;

    convert_and_norm<<<2 * NROWS, 256, 0, stream>>>(x, y, xb, yb, nx, ny);

    dim3 grid(NROWS / 128, NROWS / 128);
    cosine_gemm<<<grid, 256, 0, stream>>>(xb, yb, nx, ny, out);
}
